// TimeVariantFilter_22170621182385
// MI455X (gfx1250) — matrix operand order repaired and checked by reading, not compiled, not run on hardware
//
#include <hip/hip_runtime.h>
#include <hip/hip_bf16.h>
#include <math.h>

typedef __attribute__((ext_vector_type(8)))  _Float16 v8h;
typedef __attribute__((ext_vector_type(16))) _Float16 v16h;
typedef __attribute__((ext_vector_type(8)))  float    v8f;

#define NUM_BINS 257
#define F_PAD    288            // K padded to 9*32 for 16x16x32 WMMA
#define NUM_CH   4
#define T_REC    11
#define B_SP     8
#define G_SP     80
#define NB       8
#define TT       1500
#define PADT     5
#define ROWS     (NB*TT*B_SP)   // 96000 GEMM rows
#define M_TILES  (ROWS/16)      // 6000
#define G_TILES  (G_SP/16)      // 5
#define K_TILES  (F_PAD/32)     // 9
#define GEMM_WPB 8              // waves per block (wave32)
#define GEMM_BLOCKS (M_TILES/GEMM_WPB)  // 750

// workspace layout (bytes)
#define WS_PARTIAL 0u                       // 750*16 floats = 48000 B
#define WS_STATS   65536u                   // 16 floats: mean[8], gamma*rsqrt[8]
#define WS_WP16    131072u                  // 80*288 f16 = 46080 B
#define WS_MAG     262144u                  // 96000*288 f16 = 55.3 MB

// ---------------- init: convert/pad w_proj to f16, layout [g][k_pad] ----------
__global__ void k_init(const float* __restrict__ w_proj, _Float16* __restrict__ wp16) {
    int i = blockIdx.x * 256 + threadIdx.x;
    if (i >= G_SP * F_PAD) return;
    int g = i / F_PAD, k = i - g * F_PAD;
    float v = (k < NUM_BINS) ? w_proj[g * NUM_BINS + k] : 0.f;
    wp16[i] = (_Float16)v;
}

// ---------------- phase 1: grouped complex conv + magnitude ------------------
// block = (f, n); stages the bin's 4-channel (re,im) time series in LDS with
// halo padding, weights in LDS; each thread produces (t,b) outputs as f16 into
// mag[row][k] with row=(n*T+t)*8+b, k=f.  f in [257,288) writes the zero pad.
__global__ __launch_bounds__(256) void k_conv(
    const float* __restrict__ xr, const float* __restrict__ xi,
    const float* __restrict__ wr, const float* __restrict__ wi,
    _Float16* __restrict__ mag) {
    const int f = blockIdx.x;      // 0..287
    const int n = blockIdx.y;      // 0..7
    const int tid = threadIdx.x;

    if (f >= NUM_BINS) {           // zero-fill K padding columns
        for (int j = tid; j < TT * B_SP; j += 256) {
            int t = j >> 3, b = j & 7;
            size_t row = (size_t)(n * TT + t) * B_SP + b;
            mag[row * F_PAD + f] = (_Float16)0.f;
        }
        return;
    }

    __shared__ float shx[2][NUM_CH][TT + 2 * PADT];   // 48.3 KB
    __shared__ float shw[2][B_SP * T_REC * NUM_CH];   // 2*352 floats

    for (int idx = tid; idx < 2 * NUM_CH * (TT + 2 * PADT); idx += 256) {
        int plane = idx / (NUM_CH * (TT + 2 * PADT));
        int rem   = idx - plane * (NUM_CH * (TT + 2 * PADT));
        int c  = rem / (TT + 2 * PADT);
        int tt = rem - c * (TT + 2 * PADT);
        int ts = tt - PADT;
        const float* src = plane ? xi : xr;
        float v = 0.f;
        if (ts >= 0 && ts < TT)
            v = src[(((size_t)n * NUM_CH + c) * NUM_BINS + f) * TT + ts];
        shx[plane][c][tt] = v;
    }
    for (int idx = tid; idx < B_SP * T_REC * NUM_CH; idx += 256) {
        shw[0][idx] = wr[(size_t)f * (B_SP * T_REC * NUM_CH) + idx];
        shw[1][idx] = wi[(size_t)f * (B_SP * T_REC * NUM_CH) + idx];
    }
    __syncthreads();

    for (int j = tid; j < TT * B_SP; j += 256) {
        int t = j >> 3, b = j & 7;
        const float* wrp = &shw[0][b * T_REC * NUM_CH];
        const float* wip = &shw[1][b * T_REC * NUM_CH];
        float br = 0.f, bim = 0.f;
        #pragma unroll
        for (int tau = 0; tau < T_REC; ++tau) {
            #pragma unroll
            for (int c = 0; c < NUM_CH; ++c) {
                float xrv = shx[0][c][t + tau];
                float xiv = shx[1][c][t + tau];
                float wrv = wrp[tau * NUM_CH + c];
                float wiv = wip[tau * NUM_CH + c];
                br  = fmaf(xrv, wrv, br);
                br  = fmaf(-xiv, wiv, br);
                bim = fmaf(xiv, wrv, bim);
                bim = fmaf(xrv, wiv, bim);
            }
        }
        float m = sqrtf(fmaf(br, br, fmaf(bim, bim, 1e-5f)));
        size_t row = (size_t)(n * TT + t) * B_SP + b;
        mag[row * F_PAD + f] = (_Float16)m;
    }
}

// ---------------- phase 2: WMMA GEMM (96000x288)x(288x80) + log + partial sums
// one wave per 16-row M tile, 5 G tiles per wave (A reuse).  Accumulator VGPR
// index v corresponds exactly to spatial filter b (row%8==v), so per-b BN
// statistics reduce along accumulator lanes.
__global__ __launch_bounds__(256) void k_gemm(
    const _Float16* __restrict__ mag, const _Float16* __restrict__ wp16,
    float* __restrict__ out, float* __restrict__ partial) {
    const int tid  = threadIdx.x;
    const int wave = tid >> 5;
    const int lane = tid & 31;
    const int m_tile = blockIdx.x * GEMM_WPB + wave;   // 0..5999
    const int m_base = m_tile * 16;
    const int lrow = lane & 15;
    const int kh   = lane >> 4;   // which K-half this lane holds

    v8f acc[G_TILES];
    #pragma unroll
    for (int gt = 0; gt < G_TILES; ++gt) acc[gt] = (v8f)(0.f);

    const _Float16* arowp = mag + (size_t)(m_base + lrow) * F_PAD;

    for (int kt = 0; kt < K_TILES; ++kt) {
        const int k0 = kt * 32;
        // A 16x32 f16 layout: lane<16 rows 0-15 / elems 0-7 = K kh*16+0..7,
        // elems 8-15 = K kh*16+8..15 (matches B's K order)
        v8h alo = *(const v8h*)(arowp + k0 + kh * 16);
        v8h ahi = *(const v8h*)(arowp + k0 + kh * 16 + 8);
        v16h a;
        #pragma unroll
        for (int i = 0; i < 8; ++i) { a[i] = alo[i]; a[8 + i] = ahi[i]; }
        #pragma unroll
        for (int gt = 0; gt < G_TILES; ++gt) {
            // B 32x16: lane = column g, elems 0..15 = K kh*16+0..15
            v16h bf = *(const v16h*)(wp16 + (size_t)(gt * 16 + lrow) * F_PAD
                                     + k0 + kh * 16);
            acc[gt] = __builtin_amdgcn_wmma_f32_16x16x32_f16(
                false, a, false, bf, (short)0, acc[gt], false, false);
        }
    }

    // epilogue: y = log(relu(x)+eps); accumulate per-b (== per-v) sums
    float s[8], q[8];
    #pragma unroll
    for (int v = 0; v < 8; ++v) { s[v] = 0.f; q[v] = 0.f; }
    #pragma unroll
    for (int gt = 0; gt < G_TILES; ++gt) {
        #pragma unroll
        for (int v = 0; v < 8; ++v) {
            float y = logf(fmaxf(acc[gt][v], 0.f) + 1e-5f);
            int row = m_base + v + 8 * kh;            // C layout: M = v + 8*half
            int g   = gt * 16 + lrow;                 //           N = lane&15
            out[(size_t)row * G_SP + g] = y;
            s[v] += y; q[v] += y * y;
        }
    }
    // deterministic wave32 butterfly reduction
    #pragma unroll
    for (int v = 0; v < 8; ++v) {
        #pragma unroll
        for (int msk = 16; msk >= 1; msk >>= 1) {
            s[v] += __shfl_xor(s[v], msk, 32);
            q[v] += __shfl_xor(q[v], msk, 32);
        }
    }
    __shared__ float shred[GEMM_WPB][16];
    if (lane == 0) {
        #pragma unroll
        for (int v = 0; v < 8; ++v) { shred[wave][v] = s[v]; shred[wave][8 + v] = q[v]; }
    }
    __syncthreads();
    if (tid < 16) {
        float a2 = 0.f;
        #pragma unroll
        for (int w = 0; w < GEMM_WPB; ++w) a2 += shred[w][tid];
        partial[blockIdx.x * 16 + tid] = a2;   // full overwrite: no init needed
    }
}

// ---------------- phase 3: fixed-order final stats (deterministic) ----------
__global__ void k_stats(const float* __restrict__ partial,
                        const float* __restrict__ gamma,
                        float* __restrict__ stats) {
    int b = threadIdx.x;
    if (b >= 8) return;
    float s = 0.f, q = 0.f;
    for (int blk = 0; blk < GEMM_BLOCKS; ++blk) {
        s += partial[blk * 16 + b];
        q += partial[blk * 16 + 8 + b];
    }
    const float cnt = (float)(NB * TT * G_SP);   // 960000 per b
    float mean = s / cnt;
    float var  = q / cnt - mean * mean;
    stats[b]     = mean;
    stats[8 + b] = gamma[b] / sqrtf(var + 1e-5f);
}

// ---------------- phase 4: apply BN in place on d_out ------------------------
__global__ __launch_bounds__(256) void k_bn(float* __restrict__ out,
                                            const float* __restrict__ stats,
                                            const float* __restrict__ beta,
                                            int total) {
    int i = blockIdx.x * 256 + threadIdx.x;
    if (i >= total) return;
    int b = (i / G_SP) & 7;                      // row%8
    out[i] = (out[i] - stats[b]) * stats[8 + b] + beta[b];
}

extern "C" void kernel_launch(void* const* d_in, const int* in_sizes, int n_in,
                              void* d_out, int out_size, void* d_ws, size_t ws_size,
                              hipStream_t stream) {
    const float* xr     = (const float*)d_in[0];
    const float* xi     = (const float*)d_in[1];
    const float* wr     = (const float*)d_in[2];
    const float* wi     = (const float*)d_in[3];
    const float* w_proj = (const float*)d_in[4];
    const float* gamma  = (const float*)d_in[5];
    const float* beta   = (const float*)d_in[6];
    float* out = (float*)d_out;

    char* ws = (char*)d_ws;
    float*     partial = (float*)(ws + WS_PARTIAL);
    float*     stats   = (float*)(ws + WS_STATS);
    _Float16*  wp16    = (_Float16*)(ws + WS_WP16);
    _Float16*  mag     = (_Float16*)(ws + WS_MAG);

    k_init<<<(G_SP * F_PAD + 255) / 256, 256, 0, stream>>>(w_proj, wp16);

    dim3 gconv(F_PAD, NB);
    k_conv<<<gconv, 256, 0, stream>>>(xr, xi, wr, wi, mag);

    k_gemm<<<GEMM_BLOCKS, 256, 0, stream>>>(mag, wp16, out, partial);

    k_stats<<<1, 32, 0, stream>>>(partial, gamma, stats);

    int total = ROWS * G_SP;   // 7,680,000
    k_bn<<<(total + 255) / 256, 256, 0, stream>>>(out, stats, beta, total);
}
